// TwoBERTsModel_50130858279010
// MI455X (gfx1250) — compile-verified, matrix operand order checked by reading, not run on hardware
//
#include <hip/hip_runtime.h>
#include <hip/hip_bf16.h>

// ---------------------------------------------------------------------------
// TwoBERTs retrieval+rerank for MI455X (gfx1250, wave32, WMMA).
//
// Pipeline:
//   K1 convert  : retriever hidden/emb f32 -> bf16 (ws)
//   K2 gemm     : scores[B,N] = rh_bf16 @ re_bf16^T via v_wmma_f32_16x16x32_bf16
//                 LDS double-buffered; staging via GLOBAL_LOAD_ASYNC_TO_LDS_B128
//                 (ASYNCcnt-pipelined).
//   K3 hist     : per-row 4096-bin histogram of monotonic-key top-12 bits
//   K4 thresh   : per-row threshold bin + #items to take from boundary bin
//   K5 rerank   : per-row ordered scan; deterministic exact-K selection;
//                 selected -> sigmoid(dot(reranker_emb[n], reranker_hidden[b]))
//                 else 0; written in place over the scores.
// ---------------------------------------------------------------------------

typedef __attribute__((ext_vector_type(16))) __bf16 v16bf;
typedef __attribute__((ext_vector_type(8)))  float  v8f;

union AFrag { v16bf v; uint4 q[2]; };

#if defined(__HIP_DEVICE_COMPILE__) &&                                   \
    __has_builtin(__builtin_amdgcn_global_load_async_to_lds_b128) &&     \
    __has_builtin(__builtin_amdgcn_s_wait_asynccnt)
#define GEMM_ASYNC_STAGING 1
#else
#define GEMM_ASYNC_STAGING 0
#endif

#if GEMM_ASYNC_STAGING
// Types per the builtin's prototype (from the compiler diagnostic):
//   param0: v4i in global address space, param1: v4i in LDS address space.
typedef int tdm_v4i __attribute__((vector_size(4 * sizeof(int))));
typedef __attribute__((address_space(1))) tdm_v4i glob_v4i;
typedef __attribute__((address_space(3))) tdm_v4i shar_v4i;
#endif

static __device__ __forceinline__ unsigned short f2bf(float f) {
  unsigned u = __float_as_uint(f);
  unsigned r = u + 0x7FFFu + ((u >> 16) & 1u);   // round-to-nearest-even
  return (unsigned short)(r >> 16);
}

static __device__ __forceinline__ unsigned monokey(float f) {
  unsigned u = __float_as_uint(f);
  return (u & 0x80000000u) ? ~u : (u | 0x80000000u);
}

// ---------------------------------------------------------------- K1 convert
__global__ void k_convert(const float* __restrict__ rh, const float* __restrict__ re,
                          unsigned short* __restrict__ rh_b, unsigned short* __restrict__ re_b,
                          long long nrh, long long nre) {
  long long total = nrh + nre;
  long long stride = (long long)gridDim.x * blockDim.x;
  for (long long i = (long long)blockIdx.x * blockDim.x + threadIdx.x; i < total; i += stride) {
    if (i < nrh) rh_b[i] = f2bf(rh[i]);
    else         re_b[i - nrh] = f2bf(re[i - nrh]);
  }
}

// ------------------------------------------------------------------- K2 gemm
// 256 threads = 8 waves. Output tile 128(m) x 128(n), K=256 walked in four
// 64-deep stages through a double-buffered LDS tile (2 x 128 x 72 bf16).
// Each wave owns 16 m-rows and all 8 n-subtiles (8 v8f accumulators).
__global__ __launch_bounds__(256) void k_gemm(const unsigned short* __restrict__ rh_b,
                                              const unsigned short* __restrict__ re_b,
                                              float* __restrict__ out, int N) {
  constexpr int E  = 256;
  constexpr int KS = 64;                         // k per stage
  __shared__ unsigned short lds[2][128][72];     // +8 pad: row stride 144B

  const int tile_n  = blockIdx.x * 128;
  const int m_base0 = blockIdx.y * 128;
  const int tid  = threadIdx.x;
  const int lane = tid & 31;
  const int wave = tid >> 5;
  const int m_base = m_base0 + wave * 16;
  const int lbase  = (lane & 16) ? 8 : 0;        // per-lane K base within 32-chunk

  v8f acc[8] = {};

  // staging map: 2 threads per item row, 32 elems (4 x b128) each
  const int r_ld = tid >> 1;                     // 0..127
  const int half = (tid & 1) * 32;               // 0 or 32
  int gn = tile_n + r_ld;
  if (gn >= N) gn = N - 1;                       // clamp tail tile (never stored)
  const unsigned short* gsrc = re_b + (size_t)gn * E + half;

  auto stage = [&](int ks, int buf) {
#if GEMM_ASYNC_STAGING
#pragma unroll
    for (int i = 0; i < 4; i++)
      __builtin_amdgcn_global_load_async_to_lds_b128(
          (glob_v4i*)(gsrc + ks + i * 8),
          (shar_v4i*)&lds[buf][r_ld][half + i * 8],
          0, 0);
#else
    const uint4* g = (const uint4*)(gsrc + ks);
#pragma unroll
    for (int i = 0; i < 4; i++)
      *(uint4*)&lds[buf][r_ld][half + i * 8] = g[i];
#endif
  };

  stage(0, 0);                                   // prime the pipeline
  for (int s = 0; s < E / KS; s++) {
    const int buf = s & 1;
    if (s + 1 < E / KS) {
      stage((s + 1) * KS, buf ^ 1);              // prefetch next stage
#if GEMM_ASYNC_STAGING
      __builtin_amdgcn_s_wait_asynccnt(4);       // this wave's stage-s loads done
#endif
    } else {
#if GEMM_ASYNC_STAGING
      __builtin_amdgcn_s_wait_asynccnt(0);
#endif
    }
    __syncthreads();                             // all waves' stage-s data in LDS

#pragma unroll
    for (int kc = 0; kc < KS; kc += 32) {
      // A fragment (16x32 bf16) straight from global (rh is tiny & cache-hot)
      AFrag a;
      const unsigned short* ap =
          rh_b + (size_t)(m_base + (lane & 15)) * E + s * KS + kc + lbase;
      a.q[0] = *(const uint4*)ap;
      a.q[1] = *(const uint4*)(ap + 16);
#pragma unroll
      for (int nt = 0; nt < 8; nt++) {
        AFrag b;
        const unsigned short* bp = &lds[buf][nt * 16 + (lane & 15)][kc + lbase];
        b.q[0] = *(const uint4*)bp;
        b.q[1] = *(const uint4*)(bp + 16);
        acc[nt] = __builtin_amdgcn_wmma_f32_16x16x32_bf16(
            false, a.v, false, b.v, (short)0, acc[nt], false, false);
      }
    }
    __syncthreads();                             // before buf is restaged
  }

  // epilogue: C/D layout -> lane (&15) = n, VGPR r = m (+8 for hi half-wave)
  const int mrow_off = (lane >> 4) << 3;
#pragma unroll
  for (int nt = 0; nt < 8; nt++) {
    int n = tile_n + nt * 16 + (lane & 15);
    if (n < N) {
      float* op = out + (size_t)(m_base + mrow_off) * N + n;
#pragma unroll
      for (int r = 0; r < 8; r++) op[(size_t)r * N] = acc[nt][r];
    }
  }
}

// ------------------------------------------------------------------- K3 hist
__global__ __launch_bounds__(256) void k_hist(const float* __restrict__ scores,
                                              unsigned* __restrict__ hist, int N) {
  const int row = blockIdx.x;
  __shared__ unsigned h[4096];
  for (int i = threadIdx.x; i < 4096; i += 256) h[i] = 0u;
  __syncthreads();
  const float* s = scores + (size_t)row * N;
  for (int n = threadIdx.x; n < N; n += 256)
    atomicAdd(&h[monokey(s[n]) >> 20], 1u);
  __syncthreads();
  for (int i = threadIdx.x; i < 4096; i += 256)
    hist[(size_t)row * 4096 + i] = h[i];
}

// ----------------------------------------------------------------- K4 thresh
__global__ __launch_bounds__(256) void k_thresh(const unsigned* __restrict__ hist,
                                                int2* __restrict__ rowinfo,
                                                const int* __restrict__ Kptr) {
  const int row = blockIdx.x;
  const int tid = threadIdx.x;
  const unsigned K = (unsigned)*Kptr;
  __shared__ unsigned s_sum[256];
  const unsigned* h = hist + (size_t)row * 4096;

  const int base = 4095 - tid * 16;          // each thread: 16 bins, descending
  unsigned loc[16];
  unsigned sum = 0;
#pragma unroll
  for (int i = 0; i < 16; i++) { loc[i] = h[base - i]; sum += loc[i]; }
  s_sum[tid] = sum;
  __syncthreads();
  if (tid == 0) {
    unsigned run = 0;
    for (int t = 0; t < 256; t++) { unsigned v = s_sum[t]; s_sum[t] = run; run += v; }
  }
  __syncthreads();
  const unsigned before = s_sum[tid];
  if (before < K && before + sum >= K) {     // unique crossing thread
    unsigned run = before;
    int bin = 0; unsigned rem = 0;
#pragma unroll
    for (int i = 0; i < 16; i++) {
      if (run + loc[i] >= K) { bin = base - i; rem = K - run; break; }
      run += loc[i];
    }
    rowinfo[row] = make_int2(bin, (int)rem);
  }
}

// ----------------------------------------------------------------- K5 rerank
// One 256-thread WG per row, chunks scanned in index order so boundary-bin
// ties resolve deterministically by lowest index (stable, like top_k).
__global__ __launch_bounds__(256) void k_rerank(float* __restrict__ out,
                                                const float* __restrict__ rer_emb,
                                                const float* __restrict__ rer_hid,
                                                const int2* __restrict__ rowinfo, int N) {
  constexpr int E = 256;
  const int row = blockIdx.x;
  const int tid = threadIdx.x;
  const int lane = tid & 31;
  const int wave = tid >> 5;

  const int2 info = rowinfo[row];
  const unsigned tbin = (unsigned)info.x;
  const unsigned remK = (unsigned)info.y;

  // reranker_hidden row cached in registers: 8 floats / lane
  const float* hrow = rer_hid + (size_t)row * E;
  const float4 h0 = *(const float4*)(hrow + lane * 8);
  const float4 h1 = *(const float4*)(hrow + lane * 8 + 4);

  __shared__ unsigned s_wcnt[8];
  __shared__ unsigned s_base;
  if (tid == 0) s_base = 0u;
  __syncthreads();

  float* rowp = out + (size_t)row * N;
  const int chunks = (N + 255) / 256;
  for (int c = 0; c < chunks; c++) {
    const int n = c * 256 + tid;
    const bool valid = n < N;
    const float sc = valid ? rowp[n] : 0.0f;
    const unsigned key = valid ? (monokey(sc) >> 20) : 0u;
    const bool hi = valid && key > tbin;
    const bool eq = valid && key == tbin;

    // deterministic rank of boundary-bin hits (index order)
    const unsigned eqm = (unsigned)__ballot(eq);
    if (lane == 0) s_wcnt[wave] = (unsigned)__popc(eqm);
    __syncthreads();
    if (tid == 0) {
      unsigned run = s_base;
      for (int w = 0; w < 8; w++) { unsigned v = s_wcnt[w]; s_wcnt[w] = run; run += v; }
      s_base = run;
    }
    __syncthreads();
    const unsigned rank = s_wcnt[wave] + (unsigned)__popc(eqm & ((1u << lane) - 1u));
    const bool sel = hi || (eq && rank < remK);

    if (valid && !sel) rowp[n] = 0.0f;

    // wave-cooperative rerank dot for each selected lane
    unsigned selm = (unsigned)__ballot(sel);
    while (selm) {
      const int src = __builtin_ctz(selm);
      selm &= selm - 1u;
      const int nn = __shfl(n, src, 32);
      const float4* ep = (const float4*)(rer_emb + (size_t)nn * E);
      const float4 e0 = ep[lane * 2];
      const float4 e1 = ep[lane * 2 + 1];
      float p = e0.x * h0.x + e0.y * h0.y + e0.z * h0.z + e0.w * h0.w
              + e1.x * h1.x + e1.y * h1.y + e1.z * h1.z + e1.w * h1.w;
      for (int off = 16; off; off >>= 1) p += __shfl_xor(p, off, 32);
      if (lane == src) rowp[nn] = 1.0f / (1.0f + __expf(-p));
    }
    __syncthreads();
  }
}

// ---------------------------------------------------------------------------
extern "C" void kernel_launch(void* const* d_in, const int* in_sizes, int n_in,
                              void* d_out, int out_size, void* d_ws, size_t ws_size,
                              hipStream_t stream) {
  (void)n_in; (void)out_size; (void)ws_size;
  const float* retr_hid = (const float*)d_in[0];
  const float* rank_hid = (const float*)d_in[1];
  const float* retr_emb = (const float*)d_in[2];
  const float* rank_emb = (const float*)d_in[3];
  const int*   Kptr     = (const int*)d_in[4];

  const int E = 256;
  const int B = in_sizes[0] / E;   // 1024
  const int N = in_sizes[2] / E;   // 100000

  char* ws = (char*)d_ws;
  size_t o = 0;
  unsigned short* rh_b = (unsigned short*)(ws + o);
  o += (size_t)B * E * 2;  o = (o + 255) & ~(size_t)255;
  unsigned short* re_b = (unsigned short*)(ws + o);
  o += (size_t)N * E * 2;  o = (o + 255) & ~(size_t)255;
  unsigned* hist = (unsigned*)(ws + o);
  o += (size_t)B * 4096 * 4; o = (o + 255) & ~(size_t)255;
  int2* rowinfo = (int2*)(ws + o);

  float* out = (float*)d_out;

  k_convert<<<2048, 256, 0, stream>>>(retr_hid, retr_emb, rh_b, re_b,
                                      (long long)B * E, (long long)N * E);
  dim3 g2((N + 127) / 128, B / 128);
  k_gemm<<<g2, 256, 0, stream>>>(rh_b, re_b, out, N);
  k_hist<<<B, 256, 0, stream>>>(out, hist, N);
  k_thresh<<<B, 256, 0, stream>>>(hist, rowinfo, Kptr);
  k_rerank<<<B, 256, 0, stream>>>(out, rank_emb, rank_hid, rowinfo, N);
}
